// EGNN_dynamics_AD2_cat_86646670230064
// MI455X (gfx1250) — matrix operand order repaired and checked by reading, not compiled, not run on hardware
//
#include <hip/hip_runtime.h>
#include <hip/hip_bf16.h>
#include <math.h>

typedef __attribute__((ext_vector_type(16))) _Float16 v16h;
typedef __attribute__((ext_vector_type(8)))  _Float16 h8;
typedef __attribute__((ext_vector_type(8)))  float    v8f;

#define NPART 22
#define HIDD  64
#define NLAYERS 4
#define EPAD 464          // 462 real edges padded to 29*16
#define NTILE_E 29

// ---------------- workspace (d_ws) layout: f16 transposed weights [N][Kpad] ----
constexpr int E1_H = 64*160;   // edge_w1: K=130 pad 160
constexpr int E2_H = 64*64;
constexpr int N1_H = 64*128;
constexpr int N2_H = 64*64;
constexpr int C1_H = 64*64;
constexpr int WS_E1 = 0;
constexpr int WS_E2 = WS_E1 + 4*E1_H;
constexpr int WS_N1 = WS_E2 + 4*E2_H;
constexpr int WS_N2 = WS_N1 + 4*N1_H;
constexpr int WS_C1 = WS_N2 + 4*N2_H;
constexpr int WS_TOT = WS_C1 + 4*C1_H;   // 122880 halves = 240 KB

// ---------------- LDS layout (byte offsets, all 16B aligned) ------------------
constexpr int L_E1  = 0;
constexpr int L_E2  = L_E1  + 2*E1_H;        // 20480
constexpr int L_N1  = L_E2  + 2*E2_H;        // 28672
constexpr int L_N2  = L_N1  + 2*N1_H;        // 45056
constexpr int L_C1  = L_N2  + 2*N2_H;        // 53248
constexpr int L_H16 = L_C1  + 2*C1_H;        // 61440 : h f16 [24][72]
constexpr int L_M16 = L_H16 + 24*72*2;       // 64896 : m f16 [464][72]
constexpr int L_NIN = L_M16 + EPAD*72*2;     // 131712: node in f16 [32][136]
constexpr int L_ST  = L_NIN + 32*136*2;      // 140416: per-wave scratch [8][16][72] f16
constexpr int L_H32 = L_ST  + 8*16*72*2;     // 158848: h f32 [22][64]
constexpr int L_X   = L_H32 + 22*64*4;       // 164480: x f32 [24][3]
constexpr int L_X0  = L_X   + 24*3*4;        // 164768
constexpr int L_DN  = L_X0  + 24*3*4;        // 165056: dnorm [464][3] f32
constexpr int L_RAD = L_DN  + EPAD*3*4;      // 170624
constexpr int L_EA  = L_RAD + EPAD*4;        // 172480
constexpr int L_PHI = L_EA  + EPAD*4;        // 174336
constexpr int L_MEAN= L_PHI + EPAD*4;        // 176192
constexpr int L_RID = L_MEAN+ 16;            // 176208: u16 [464]
constexpr int L_CID = L_RID + EPAD*2;        // 177136
constexpr int L_NEDG= L_CID + EPAD*2;        // 178064: u16 [22][24]
constexpr int L_NCNT= L_NEDG+ 22*24*2;       // 179120: int [24]
constexpr int SMEM_BYTES = L_NCNT + 24*4;    // 179216 B  (< 320 KB/WG)

// ---------------- device helpers ---------------------------------------------
__device__ inline v8f wmma16(v16h a, v16h b, v8f c) {
  return __builtin_amdgcn_wmma_f32_16x16x32_f16(false, a, false, b, (short)0, c, false, false);
}
// A fragment (16x32 f16): lane holds row m=lane%16, K = kbase+{0..7} and kbase+16+{0..7}
__device__ inline v16h afrag(const _Float16* row, int c0) {
  h8 x = *(const h8*)(row + c0);
  h8 y = *(const h8*)(row + c0 + 16);
  v16h a;
#pragma unroll
  for (int i = 0; i < 8; ++i) { a[i] = x[i]; a[i + 8] = y[i]; }
  return a;
}
// B fragment (32x16 f16): lane holds col n, K = kt*32 + hi*8 + {0..7} and kt*32 + 16 + hi*8 + {0..7}
__device__ inline v16h bfrag(const _Float16* wT, int strideK, int col, int kt, int hi) {
  const _Float16* q = wT + col * strideK + kt * 32 + hi * 8;
  h8 x = *(const h8*)q;
  h8 y = *(const h8*)(q + 16);
  v16h b;
#pragma unroll
  for (int i = 0; i < 8; ++i) { b[i] = x[i]; b[i + 8] = y[i]; }
  return b;
}
// fast SiLU: v * rcp(1+exp(-v))  -> v_exp_f32 + v_rcp_f32 (TRANS, co-exec with WMMA)
__device__ inline float silu_f(float v) {
  return v * __builtin_amdgcn_rcpf(1.f + __expf(-v));
}
// accumulator pre-loaded with the per-column bias (D column is fixed per lane)
__device__ inline v8f vbias8(float b) {
  v8f z;
#pragma unroll
  for (int i = 0; i < 8; ++i) z[i] = b;
  return z;
}

// ---------------- prep: convert weights to f16, transposed [N][Kpad] ----------
__global__ void prep_weights(const float* __restrict__ e1, const float* __restrict__ e2,
                             const float* __restrict__ n1, const float* __restrict__ n2,
                             const float* __restrict__ c1, _Float16* __restrict__ ws) {
  int idx = blockIdx.x * 256 + threadIdx.x;
  if (idx < WS_E2) {
    int t = idx - WS_E1; int l = t / E1_H; int r = t % E1_H; int n = r / 160; int k = r % 160;
    ws[idx] = (_Float16)((k < 130) ? e1[(l * 130 + k) * 64 + n] : 0.f);
  } else if (idx < WS_N1) {
    int t = idx - WS_E2; int l = t / E2_H; int r = t % E2_H; int n = r / 64; int k = r % 64;
    ws[idx] = (_Float16)e2[(l * 64 + k) * 64 + n];
  } else if (idx < WS_N2) {
    int t = idx - WS_N1; int l = t / N1_H; int r = t % N1_H; int n = r / 128; int k = r % 128;
    ws[idx] = (_Float16)n1[(l * 128 + k) * 64 + n];
  } else if (idx < WS_C1) {
    int t = idx - WS_N2; int l = t / N2_H; int r = t % N2_H; int n = r / 64; int k = r % 64;
    ws[idx] = (_Float16)n2[(l * 64 + k) * 64 + n];
  } else if (idx < WS_TOT) {
    int t = idx - WS_C1; int l = t / C1_H; int r = t % C1_H; int n = r / 64; int k = r % 64;
    ws[idx] = (_Float16)c1[(l * 64 + k) * 64 + n];
  }
}

// ---------------- fused EGNN: one workgroup per batch sample ------------------
__global__ void __launch_bounds__(256)
egnn_fused(const float* __restrict__ tg, const float* __restrict__ xs,
           const float* __restrict__ h_init, const float* __restrict__ emb_w,
           const float* __restrict__ emb_b,
           const float* __restrict__ eb1g, const float* __restrict__ eb2g,
           const float* __restrict__ nb1g, const float* __restrict__ nb2g,
           const float* __restrict__ cb1g, const float* __restrict__ c2g,
           const _Float16* __restrict__ ws, float* __restrict__ outp) {
  extern __shared__ char smem[];
  _Float16* e1T = (_Float16*)(smem + L_E1);
  _Float16* e2T = (_Float16*)(smem + L_E2);
  _Float16* n1T = (_Float16*)(smem + L_N1);
  _Float16* n2T = (_Float16*)(smem + L_N2);
  _Float16* c1T = (_Float16*)(smem + L_C1);
  _Float16* h16 = (_Float16*)(smem + L_H16);
  _Float16* m16 = (_Float16*)(smem + L_M16);
  _Float16* nin = (_Float16*)(smem + L_NIN);
  _Float16* st  = (_Float16*)(smem + L_ST);
  float* h32  = (float*)(smem + L_H32);
  float* xsd  = (float*)(smem + L_X);
  float* x0s  = (float*)(smem + L_X0);
  float* dn   = (float*)(smem + L_DN);
  float* rad  = (float*)(smem + L_RAD);
  float* ea   = (float*)(smem + L_EA);
  float* phix = (float*)(smem + L_PHI);
  float* meanb= (float*)(smem + L_MEAN);
  unsigned short* rid  = (unsigned short*)(smem + L_RID);
  unsigned short* cid  = (unsigned short*)(smem + L_CID);
  unsigned short* nedg = (unsigned short*)(smem + L_NEDG);
  int* ncnt = (int*)(smem + L_NCNT);

  const int b    = blockIdx.x;
  const int tid  = threadIdx.x;
  const int lane = tid & 31;
  const int wave = tid >> 5;
  const int nlo  = lane & 15;
  const int hi   = lane >> 4;
  const int khi  = hi * 8;
  _Float16* stw = st + wave * 16 * 72;

  // ------------------ init: edge ids, x/x0, zero pads ------------------------
  if (tid < 231) {
    int rem = tid, i = 0, cnt = NPART - 1;
    while (rem >= cnt) { rem -= cnt; ++i; --cnt; }
    int j = i + 1 + rem;
    rid[2 * tid] = (unsigned short)i; cid[2 * tid] = (unsigned short)j;
    rid[2 * tid + 1] = (unsigned short)j; cid[2 * tid + 1] = (unsigned short)i;
  }
  if (tid == 231) { rid[462] = 22; cid[462] = 22; rid[463] = 22; cid[463] = 22; }
  if (tid < 24) ncnt[tid] = 0;
  for (int idx = tid; idx < 72; idx += 256) {
    float v = (idx < 66) ? xs[b * 66 + idx] : 0.f;
    x0s[idx] = v; xsd[idx] = v;
  }
  for (int idx = tid; idx < 2 * 72; idx += 256) h16[22 * 72 + idx] = (_Float16)0.f;
  for (int idx = tid; idx < 32 * 136; idx += 256) nin[idx] = (_Float16)0.f;
  __syncthreads();

  // per-node edge lists, static edge_attr, time-cat embedding
  if (tid < 462) {
    int r = rid[tid];
    int s = atomicAdd(&ncnt[r], 1);
    nedg[r * 24 + s] = (unsigned short)tid;
  }
  for (int e = tid; e < EPAD; e += 256) {
    int r = rid[e], c = cid[e];
    float dx = x0s[r * 3] - x0s[c * 3];
    float dy = x0s[r * 3 + 1] - x0s[c * 3 + 1];
    float dz = x0s[r * 3 + 2] - x0s[c * 3 + 2];
    ea[e] = dx * dx + dy * dy + dz * dz;
  }
  {
    float tval = tg[b];
    for (int idx = tid; idx < NPART * HIDD; idx += 256) {
      int n = idx >> 6, f = idx & 63;
      float a = emb_b[f];
#pragma unroll
      for (int k = 0; k < 8; ++k) a += h_init[n * 8 + k] * emb_w[k * 64 + f];
      a += tval * emb_w[8 * 64 + f];
      h32[idx] = a;
      h16[n * 72 + f] = (_Float16)a;
    }
  }
  __syncthreads();

  // ------------------------------- layers ------------------------------------
  for (int l = 0; l < NLAYERS; ++l) {
    // stage transposed f16 weights for this layer into LDS (b128 copies)
    {
      const uint4* s; uint4* d;
      s = (const uint4*)(ws + WS_E1 + l * E1_H); d = (uint4*)e1T;
      for (int i = tid; i < E1_H / 8; i += 256) d[i] = s[i];
      s = (const uint4*)(ws + WS_E2 + l * E2_H); d = (uint4*)e2T;
      for (int i = tid; i < E2_H / 8; i += 256) d[i] = s[i];
      s = (const uint4*)(ws + WS_N1 + l * N1_H); d = (uint4*)n1T;
      for (int i = tid; i < N1_H / 8; i += 256) d[i] = s[i];
      s = (const uint4*)(ws + WS_N2 + l * N2_H); d = (uint4*)n2T;
      for (int i = tid; i < N2_H / 8; i += 256) d[i] = s[i];
      s = (const uint4*)(ws + WS_C1 + l * C1_H); d = (uint4*)c1T;
      for (int i = tid; i < C1_H / 8; i += 256) d[i] = s[i];
    }
    // per-edge radial / normalized diff from current x
    for (int e = tid; e < EPAD; e += 256) {
      int r = rid[e], c = cid[e];
      float dx = xsd[r * 3] - xsd[c * 3];
      float dy = xsd[r * 3 + 1] - xsd[c * 3 + 1];
      float dz = xsd[r * 3 + 2] - xsd[c * 3 + 2];
      float rr = dx * dx + dy * dy + dz * dz;
      rad[e] = rr;
      float inv = __builtin_amdgcn_rcpf(sqrtf(rr) + 1.f);
      dn[e * 3] = dx * inv; dn[e * 3 + 1] = dy * inv; dn[e * 3 + 2] = dz * inv;
    }
    __syncthreads();

    const float* eb1 = eb1g + l * 64;
    const float* eb2 = eb2g + l * 64;
    const float* cb1 = cb1g + l * 64;
    const float* c2w = c2g + l * 64;

    // -------- edge phase: each wave owns edge tiles (tile = 16 edges) --------
    for (int tile = wave; tile < NTILE_E; tile += 8) {
      int eb = tile * 16;
      int er = eb + nlo;
      int rn = rid[er], cn = cid[er];
      v16h A[5];
      A[0] = afrag(h16 + rn * 72, khi);
      A[1] = afrag(h16 + rn * 72, 32 + khi);
      A[2] = afrag(h16 + cn * 72, khi);
      A[3] = afrag(h16 + cn * 72, 32 + khi);
      {
        v16h a4;
#pragma unroll
        for (int i = 0; i < 16; ++i) a4[i] = (_Float16)0.f;
        if (!hi) { a4[0] = (_Float16)rad[er]; a4[1] = (_Float16)ea[er]; }
        A[4] = a4;
      }
      // edge MLP stage 1: e_in[16,160] @ W1 (+bias in acc) -> silu -> scratch
#pragma unroll
      for (int nt = 0; nt < 4; ++nt) {
        v8f acc = vbias8(eb1[nt * 16 + nlo]);
#pragma unroll
        for (int kt = 0; kt < 5; ++kt)
          acc = wmma16(A[kt], bfrag(e1T, 160, nt * 16 + nlo, kt, hi), acc);
#pragma unroll
        for (int j = 0; j < 8; ++j)
          stw[(j + khi) * 72 + nt * 16 + nlo] = (_Float16)silu_f(acc[j]);
      }
      // edge MLP stage 2: -> m (store to m16 and scratch)
      v16h A2[2];
      A2[0] = afrag(stw + nlo * 72, khi);
      A2[1] = afrag(stw + nlo * 72, 32 + khi);
      v8f acc2[4];
#pragma unroll
      for (int nt = 0; nt < 4; ++nt) {
        v8f acc = vbias8(eb2[nt * 16 + nlo]);
#pragma unroll
        for (int kt = 0; kt < 2; ++kt)
          acc = wmma16(A2[kt], bfrag(e2T, 64, nt * 16 + nlo, kt, hi), acc);
        acc2[nt] = acc;
      }
#pragma unroll
      for (int nt = 0; nt < 4; ++nt) {
#pragma unroll
        for (int j = 0; j < 8; ++j) {
          _Float16 v = (_Float16)silu_f(acc2[nt][j]);
          m16[(eb + j + khi) * 72 + nt * 16 + nlo] = v;
          stw[(j + khi) * 72 + nt * 16 + nlo] = v;
        }
      }
      // coord MLP: silu(m@c1+b1) . c2  -> phi_x per edge, then x += dnorm*phi
      v16h A3[2];
      A3[0] = afrag(stw + nlo * 72, khi);
      A3[1] = afrag(stw + nlo * 72, 32 + khi);
      float pj[8];
#pragma unroll
      for (int j = 0; j < 8; ++j) pj[j] = 0.f;
#pragma unroll
      for (int nt = 0; nt < 4; ++nt) {
        v8f acc = vbias8(cb1[nt * 16 + nlo]);
#pragma unroll
        for (int kt = 0; kt < 2; ++kt)
          acc = wmma16(A3[kt], bfrag(c1T, 64, nt * 16 + nlo, kt, hi), acc);
        float c2v = c2w[nt * 16 + nlo];
#pragma unroll
        for (int j = 0; j < 8; ++j) pj[j] += silu_f(acc[j]) * c2v;
      }
#pragma unroll
      for (int s = 1; s < 16; s <<= 1) {
#pragma unroll
        for (int j = 0; j < 8; ++j) pj[j] += __shfl_xor(pj[j], s, 32);
      }
      if (nlo == 0) {
#pragma unroll
        for (int j = 0; j < 8; ++j) phix[eb + khi + j] = pj[j];
      }
      if (lane < 16) {
        int e = eb + lane;
        float px = phix[e];
        int r2 = rid[e];
        atomicAdd(&xsd[r2 * 3 + 0], dn[e * 3 + 0] * px);
        atomicAdd(&xsd[r2 * 3 + 1], dn[e * 3 + 1] * px);
        atomicAdd(&xsd[r2 * 3 + 2], dn[e * 3 + 2] * px);
      }
    }
    __syncthreads();

    // -------- aggregate m over incoming edges, build node-MLP input ----------
    for (int idx = tid; idx < NPART * HIDD; idx += 256) {
      int n = idx >> 6, f = idx & 63;
      float s = 0.f;
      for (int k = 0; k < 21; ++k) {
        int e = nedg[n * 24 + k];
        s += (float)m16[e * 72 + f];
      }
      nin[n * 136 + f] = h16[n * 72 + f];
      nin[n * 136 + 64 + f] = (_Float16)s;
    }
    __syncthreads();

    // -------- node MLP (2 M-tiles -> waves 0,1), residual update of h --------
    if (wave < 2) {
      int rowb = wave * 16;
      v16h An[4];
#pragma unroll
      for (int kt = 0; kt < 4; ++kt) An[kt] = afrag(nin + (rowb + nlo) * 136, kt * 32 + khi);
#pragma unroll
      for (int nt = 0; nt < 4; ++nt) {
        v8f acc = vbias8(nb1g[l * 64 + nt * 16 + nlo]);
#pragma unroll
        for (int kt = 0; kt < 4; ++kt)
          acc = wmma16(An[kt], bfrag(n1T, 128, nt * 16 + nlo, kt, hi), acc);
#pragma unroll
        for (int j = 0; j < 8; ++j)
          stw[(j + khi) * 72 + nt * 16 + nlo] = (_Float16)silu_f(acc[j]);
      }
      v16h A2[2];
      A2[0] = afrag(stw + nlo * 72, khi);
      A2[1] = afrag(stw + nlo * 72, 32 + khi);
#pragma unroll
      for (int nt = 0; nt < 4; ++nt) {
        v8f acc = vbias8(nb2g[l * 64 + nt * 16 + nlo]);
#pragma unroll
        for (int kt = 0; kt < 2; ++kt)
          acc = wmma16(A2[kt], bfrag(n2T, 64, nt * 16 + nlo, kt, hi), acc);
#pragma unroll
        for (int j = 0; j < 8; ++j) {
          int M = rowb + j + khi;
          if (M < NPART) {
            float hv = h32[M * 64 + nt * 16 + nlo] + acc[j];
            h32[M * 64 + nt * 16 + nlo] = hv;
            h16[M * 72 + nt * 16 + nlo] = (_Float16)hv;
          }
        }
      }
    }
    __syncthreads();
  }

  // ---------------- output: vel = (x - x0) minus per-dim mean ----------------
  if (tid < 3) {
    float mv = 0.f;
    for (int n = 0; n < NPART; ++n) mv += xsd[n * 3 + tid] - x0s[n * 3 + tid];
    meanb[tid] = mv * (1.f / (float)NPART);
  }
  __syncthreads();
  for (int idx = tid; idx < 66; idx += 256) {
    int d = idx % 3;
    outp[b * 66 + idx] = xsd[idx] - x0s[idx] - meanb[d];
  }
}

// ------------------------------- launch ---------------------------------------
extern "C" void kernel_launch(void* const* d_in, const int* in_sizes, int n_in,
                              void* d_out, int out_size, void* d_ws, size_t ws_size,
                              hipStream_t stream) {
  (void)n_in; (void)out_size; (void)ws_size;
  const float* tg   = (const float*)d_in[0];
  const float* xs   = (const float*)d_in[1];
  const float* hini = (const float*)d_in[2];
  const float* embw = (const float*)d_in[3];
  const float* embb = (const float*)d_in[4];
  const float* ew1  = (const float*)d_in[5];
  const float* eb1  = (const float*)d_in[6];
  const float* ew2  = (const float*)d_in[7];
  const float* eb2  = (const float*)d_in[8];
  const float* nw1  = (const float*)d_in[9];
  const float* nb1  = (const float*)d_in[10];
  const float* nw2  = (const float*)d_in[11];
  const float* nb2  = (const float*)d_in[12];
  const float* cw1  = (const float*)d_in[13];
  const float* cb1  = (const float*)d_in[14];
  const float* cw2  = (const float*)d_in[15];
  float* outp = (float*)d_out;
  _Float16* ws = (_Float16*)d_ws;
  const int B = in_sizes[0];   // batch (t has BATCH elements)

  prep_weights<<<(WS_TOT + 255) / 256, 256, 0, stream>>>(ew1, ew2, nw1, nw2, cw1, ws);
  egnn_fused<<<B, 256, SMEM_BYTES, stream>>>(tg, xs, hini, embw, embb,
                                             eb1, eb2, nb1, nb2, cb1, cw2, ws, outp);
}
